// NerfModel_30829275251333
// MI455X (gfx1250) — matrix operand order repaired and checked by reading, not compiled, not run on hardware
//
#include <hip/hip_runtime.h>
#include <stdint.h>

typedef _Float16 half_t;
typedef _Float16 v16h __attribute__((ext_vector_type(16)));
typedef _Float16 v8h  __attribute__((ext_vector_type(8)));
typedef float    v8f  __attribute__((ext_vector_type(8)));

#define ROWS    128      // rows of x per workgroup
#define STR     136      // padded LDS row stride (halfs): 272B/row, 16B-aligned frags
#define THREADS 256      // 8 waves (wave32)

// LDS partition (bytes), all 16B aligned
#define OFF_HA  0                         // 128*136*2 = 34816
#define OFF_HB  34816
#define OFF_W2  69632
#define OFF_W3  104448
#define OFF_XS  139264                    // 128*8 floats = 4096
#define OFF_W1  143360                    // 128*8 floats = 4096
#define SMEM_BYTES 147456

// ---------------- weight pre-convert: fp32 -> f16 into workspace ----------------
__global__ void wcvt_kernel(const float* __restrict__ W2, const float* __restrict__ W3,
                            half_t* __restrict__ w2h, half_t* __restrict__ w3h) {
  int i = blockIdx.x * blockDim.x + threadIdx.x;
  if (i < 128 * 128) {
    w2h[i] = (half_t)W2[i];
    w3h[i] = (half_t)W3[i];
  }
}

// ---------------- WMMA fragment loaders (ISA 7.12.2 layouts) ----------------
// A (16x32 f16): lane m = lane%16; lo-lanes K {k0+0..7, k0+16..23}, hi-lanes +8
__device__ __forceinline__ v16h load_frag_a(const half_t* h, int rowb, int k0, int lane) {
  const int m  = lane & 15;
  const int h8 = (lane >> 4) << 3;           // 0 | 8
  const half_t* base = h + (rowb + m) * STR + k0 + h8;
  v8h lo = *(const v8h*)(base);              // K = k0+h8 .. +7
  v8h hi = *(const v8h*)(base + 16);         // K = k0+16+h8 .. +7
  v16h r;
#pragma unroll
  for (int i = 0; i < 8; ++i) { r[i] = lo[i]; r[i + 8] = hi[i]; }
  return r;
}

// B (32x16 f16): lane n = lane%16; lo-lanes K {k0+0..7, k0+16..23}, hi-lanes +8
__device__ __forceinline__ v16h load_frag_b(const half_t* w, int c0, int k0, int lane) {
  const int n   = lane & 15;
  const int h16 = (lane >> 4) << 3;          // 0 | 8
  const half_t* base = w + (c0 + n) * STR + k0 + h16;
  v8h lo = *(const v8h*)(base);
  v8h hi = *(const v8h*)(base + 16);
  v16h r;
#pragma unroll
  for (int i = 0; i < 8; ++i) { r[i] = lo[i]; r[i + 8] = hi[i]; }
  return r;
}

// One 128->128 layer for this wave's 16 rows: hdst = relu(hsrc @ W^T + b)
__device__ __forceinline__ void wmma_layer(const half_t* __restrict__ hsrc,
                                           half_t* __restrict__ hdst,
                                           const half_t* __restrict__ wlds,
                                           const float* __restrict__ bias,
                                           int wave, int lane) {
  const int rowb = wave * 16;
  const int n    = lane & 15;
  const int hi8  = (lane >> 4) << 3;
  // cache the wave's full A panel (16x128) in registers: 4 K-tiles
  v16h a0 = load_frag_a(hsrc, rowb,  0, lane);
  v16h a1 = load_frag_a(hsrc, rowb, 32, lane);
  v16h a2 = load_frag_a(hsrc, rowb, 64, lane);
  v16h a3 = load_frag_a(hsrc, rowb, 96, lane);
#pragma unroll
  for (int ct = 0; ct < 8; ++ct) {
    const int c0 = ct * 16;
    const float bv = bias[c0 + n];           // D-lane's column bias (same for all 8 regs)
    v8f acc = { bv, bv, bv, bv, bv, bv, bv, bv };
    acc = __builtin_amdgcn_wmma_f32_16x16x32_f16(false, a0, false, load_frag_b(wlds, c0,  0, lane), (short)0, acc, false, false);
    acc = __builtin_amdgcn_wmma_f32_16x16x32_f16(false, a1, false, load_frag_b(wlds, c0, 32, lane), (short)0, acc, false, false);
    acc = __builtin_amdgcn_wmma_f32_16x16x32_f16(false, a2, false, load_frag_b(wlds, c0, 64, lane), (short)0, acc, false, false);
    acc = __builtin_amdgcn_wmma_f32_16x16x32_f16(false, a3, false, load_frag_b(wlds, c0, 96, lane), (short)0, acc, false, false);
#pragma unroll
    for (int v = 0; v < 8; ++v) {            // D: (m = v + 8*hi, n)
      float s = acc[v];
      s = s > 0.f ? s : 0.f;
      hdst[(rowb + v + hi8) * STR + c0 + n] = (half_t)s;
    }
  }
}

// ---------------- fused MLP kernel ----------------
__launch_bounds__(THREADS, 2)
__global__ void mlp_fused_kernel(const float* __restrict__ x,
                                 const float* __restrict__ W1, const float* __restrict__ b1,
                                 const half_t* __restrict__ w2g, const float* __restrict__ b2,
                                 const half_t* __restrict__ w3g, const float* __restrict__ b3,
                                 const float* __restrict__ W4, const float* __restrict__ b4,
                                 float* __restrict__ out, int ntotal) {
  extern __shared__ char smem[];
  half_t* hA  = (half_t*)(smem + OFF_HA);
  half_t* hB  = (half_t*)(smem + OFF_HB);
  half_t* w2  = (half_t*)(smem + OFF_W2);
  half_t* w3  = (half_t*)(smem + OFF_W3);
  float*  xs  = (float*) (smem + OFF_XS);
  float*  w1s = (float*) (smem + OFF_W1);

  const int  tid  = threadIdx.x;
  const int  lane = tid & 31;
  const int  wave = tid >> 5;
  const long row0 = (long)blockIdx.x * ROWS;

  // ---- stage x tile (fp32, pad-8 rows) ----
  for (int i = tid; i < ROWS * 6; i += THREADS) {
    const int r = i / 6, c = i - r * 6;
    xs[r * 8 + c] = (row0 + r < ntotal) ? x[(row0 + r) * 6 + c] : 0.f;
  }
  // ---- stage W1 (fp32, pad-8 rows) ----
  for (int i = tid; i < 128 * 6; i += THREADS) {
    const int c = i / 6, k = i - c * 6;
    w1s[c * 8 + k] = W1[i];
  }
  // ---- stage W2h/W3h (f16) into padded LDS, 32-bit copies ----
  const uint32_t* g2 = (const uint32_t*)w2g;
  const uint32_t* g3 = (const uint32_t*)w3g;
  for (int i = tid; i < 128 * 64; i += THREADS) {
    const int r = i >> 6, c2 = i & 63;       // LDS dword idx: r*68 + c2 (STR/2 = 68)
    ((uint32_t*)w2)[r * 68 + c2] = g2[r * 64 + c2];
    ((uint32_t*)w3)[r * 68 + c2] = g3[r * 64 + c2];
  }
  __syncthreads();

  // ---- layer 1 (K=6, VALU fp32): hA = f16(relu(x @ W1^T + b1)) ----
  for (int i = tid; i < ROWS * 128; i += THREADS) {
    const int r = i >> 7, c = i & 127;
    const float* xr = xs  + r * 8;
    const float* wr = w1s + c * 8;
    float s = b1[c]
            + xr[0]*wr[0] + xr[1]*wr[1] + xr[2]*wr[2]
            + xr[3]*wr[3] + xr[4]*wr[4] + xr[5]*wr[5];
    s = s > 0.f ? s : 0.f;
    hA[r * STR + c] = (half_t)s;
  }
  __syncthreads();

  // ---- layers 2 & 3: WMMA, wave-local rows (no block barrier needed between) ----
  wmma_layer(hA, hB, w2, b2, wave, lane);    // hB = relu(hA @ W2^T + b2)
  wmma_layer(hB, hA, w3, b3, wave, lane);    // hA = relu(hB @ W3^T + b3)
  __syncthreads();

  // ---- layer 4 (128 -> 4, VALU): out = hA @ W4^T + b4 ----
  for (int i = tid; i < ROWS * 4; i += THREADS) {
    const int r = i >> 2, j = i & 3;
    if (row0 + r >= ntotal) continue;
    const half_t* hr = hA + r * STR;
    const float*  wr = W4 + j * 128;
    float s = b4[j];
#pragma unroll
    for (int k8 = 0; k8 < 16; ++k8) {
      v8h hv = *(const v8h*)(hr + k8 * 8);
      const float* w8 = wr + k8 * 8;
#pragma unroll
      for (int u = 0; u < 8; ++u) s += (float)hv[u] * w8[u];
    }
    out[(row0 + r) * 4 + j] = s;
  }
}

extern "C" void kernel_launch(void* const* d_in, const int* in_sizes, int n_in,
                              void* d_out, int out_size, void* d_ws, size_t ws_size,
                              hipStream_t stream) {
  const float* x  = (const float*)d_in[0];
  const float* W1 = (const float*)d_in[1];
  const float* b1 = (const float*)d_in[2];
  const float* W2 = (const float*)d_in[3];
  const float* b2 = (const float*)d_in[4];
  const float* W3 = (const float*)d_in[5];
  const float* b3 = (const float*)d_in[6];
  const float* W4 = (const float*)d_in[7];
  const float* b4 = (const float*)d_in[8];
  float* out = (float*)d_out;

  half_t* w2h = (half_t*)d_ws;
  half_t* w3h = w2h + 128 * 128;

  const int N = in_sizes[0] / 6;

  wcvt_kernel<<<(128 * 128 + 255) / 256, 256, 0, stream>>>(W2, W3, w2h, w3h);

  const int nblocks = (N + ROWS - 1) / ROWS;
  mlp_fused_kernel<<<nblocks, THREADS, SMEM_BYTES, stream>>>(
      x, W1, b1, w2h, b2, w3h, b3, W4, b4, out, N);
}
